// GCNLayer_59579786330157
// MI455X (gfx1250) — compile-verified, matrix operand order checked by reading, not run on hardware
//
#include <hip/hip_runtime.h>
#include <hip/hip_bf16.h>

#define NROWS 8192
#define DIM   128

typedef __attribute__((ext_vector_type(16))) __bf16 v16bf;
typedef __attribute__((ext_vector_type(8)))  __bf16 v8bf;
typedef __attribute__((ext_vector_type(8)))  float  v8f;
typedef __attribute__((ext_vector_type(2)))  float  v2f;

#if __has_builtin(__builtin_amdgcn_sched_barrier)
#define SCHED_FENCE() __builtin_amdgcn_sched_barrier(0)
#else
#define SCHED_FENCE()
#endif

// ---------------------------------------------------------------------------
// Kernel 1: d_inv[i] = rsqrt(1 + sum_j A[i][j])   (A_hat = A + I row sums)
// ---------------------------------------------------------------------------
__global__ void __launch_bounds__(256)
rowsum_rsqrt(const float* __restrict__ A, float* __restrict__ d_inv) {
    __shared__ float red[256];
    const int row = blockIdx.x;
    const float* a = A + (size_t)row * NROWS;
    float s = 0.f;
    #pragma unroll
    for (int j = 0; j < NROWS / 256; j += 4) {   // 8 float4 loads per thread
        float4 v = *(const float4*)(a + threadIdx.x * 4 + j * 256);
        s += (v.x + v.y) + (v.z + v.w);
    }
    red[threadIdx.x] = s;
    __syncthreads();
    for (int off = 128; off > 0; off >>= 1) {
        if (threadIdx.x < off) red[threadIdx.x] += red[threadIdx.x + off];
        __syncthreads();
    }
    if (threadIdx.x == 0) d_inv[row] = rsqrtf(red[0] + 1.0f);
}

// ---------------------------------------------------------------------------
// Kernel 2: XsT[c][r] = bf16( d_inv[r] * X[r][c] )  (pre-scaled, transposed)
// ---------------------------------------------------------------------------
__global__ void __launch_bounds__(256)
build_xst(const float* __restrict__ X, const float* __restrict__ d_inv,
          __bf16* __restrict__ XsT) {
    const int idx = blockIdx.x * 256 + threadIdx.x;
    const int c = idx >> 13;           // / NROWS
    const int r = idx & (NROWS - 1);
    XsT[(size_t)c * NROWS + r] = (__bf16)(d_inv[r] * X[(size_t)r * DIM + c]);
}

__device__ __forceinline__ void cvt4(v16bf& d, int base, float4 f) {
    d[base + 0] = (__bf16)f.x; d[base + 1] = (__bf16)f.y;
    d[base + 2] = (__bf16)f.z; d[base + 3] = (__bf16)f.w;
}

// One k-step's worth of operand data (A rows as f32, 8 B tiles as bf16).
struct KTile {
    float4 f0, f1, f2, f3;
    v8bf   b0[8], b1[8];
};

__device__ __forceinline__ void load_tile(KTile& t, const float* pArow,
                                          const __bf16* XsT, int kb,
                                          int kb1, int kb2, int l15) {
    t.f0 = *(const float4*)(pArow + kb + kb1);
    t.f1 = *(const float4*)(pArow + kb + kb1 + 4);
    t.f2 = *(const float4*)(pArow + kb + kb2);
    t.f3 = *(const float4*)(pArow + kb + kb2 + 4);
    #pragma unroll
    for (int n = 0; n < 8; ++n) {
        const __bf16* pB = XsT + (size_t)(n * 16 + l15) * NROWS + kb;
        t.b0[n] = *(const v8bf*)(pB + kb1);
        t.b1[n] = *(const v8bf*)(pB + kb2);
    }
}

__device__ __forceinline__ void compute_tile(const KTile& t, v8f acc[8]) {
    v16bf afrag;
    cvt4(afrag, 0, t.f0); cvt4(afrag, 4, t.f1);
    cvt4(afrag, 8, t.f2); cvt4(afrag, 12, t.f3);
    #pragma unroll
    for (int n = 0; n < 8; ++n) {
        v16bf bfrag;
        #pragma unroll
        for (int i = 0; i < 8; ++i) { bfrag[i] = t.b0[n][i]; bfrag[8 + i] = t.b1[n][i]; }
        acc[n] = __builtin_amdgcn_wmma_f32_16x16x32_bf16(
            false, afrag, false, bfrag, (short)0, acc[n], false, false);
    }
}

// ---------------------------------------------------------------------------
// Kernel 3: H = d_i * (A @ Xs + Xs)   via bf16 WMMA, f32 accumulate.
//
// Block = 256 threads (8 waves) owns a 16-row x 128-col output tile; K split
// across the 8 waves (1024 each) -> 4096 waves. Two-deep software pipeline
// with sched_barrier(0) fences so the machine scheduler CANNOT sink the
// next-step loads back down to their WMMA consumers: every B pair is issued
// one full compute block (~8 WMMAs) before use, covering L2 latency.
// A (HBM) additionally prefetched 4 k-steps ahead via global_prefetch.
// ---------------------------------------------------------------------------
__global__ void __launch_bounds__(256, 2)
spmm_wmma(const float* __restrict__ A, const float* __restrict__ X,
          const __bf16* __restrict__ XsT, const float* __restrict__ d_inv,
          float* __restrict__ H) {
    __shared__ float buf[4 * 16 * DIM];           // 32 KB
    const int lane = threadIdx.x & 31;
    const int wave = threadIdx.x >> 5;
    const int half = lane >> 4;
    const int l15  = lane & 15;
    const int r0   = blockIdx.x * 16;

    v8f acc[8] = {};

    const float* pArow = A + (size_t)(r0 + l15) * NROWS;
    const int kb1  = half * 8;        // first 8-wide K group for this lane half
    const int kb2  = 16 + half * 8;   // second 8-wide K group
    const int kBeg = wave * (NROWS / 8);
    const int kEnd = kBeg + (NROWS / 8);

    KTile tA, tB;
    load_tile(tA, pArow, XsT, kBeg, kb1, kb2, l15);

    for (int kb = kBeg; kb < kEnd; kb += 64) {     // 16 iterations, 2 steps each
        __builtin_prefetch(pArow + kb + 128, 0, 3);      // A line 4 steps ahead
        __builtin_prefetch(pArow + kb + 160, 0, 3);

        load_tile(tB, pArow, XsT, kb + 32, kb1, kb2, l15);
        SCHED_FENCE();               // loads stay above, WMMAs stay below
        compute_tile(tA, acc);

        const int kn = (kb + 64 < kEnd) ? kb + 64 : kBeg;  // tail-safe clamp
        load_tile(tA, pArow, XsT, kn, kb1, kb2, l15);
        SCHED_FENCE();
        compute_tile(tB, acc);
    }

    // ---- cross-wave K reduction (deterministic two-phase LDS) ----
    // C layout: VGPR p -> row (p + half*8), col (n*16 + l15) of the 16x16 tile.
    if (wave >= 4) {
        #pragma unroll
        for (int n = 0; n < 8; ++n)
            #pragma unroll
            for (int p = 0; p < 8; ++p)
                buf[(wave - 4) * 2048 + (p + half * 8) * DIM + n * 16 + l15] = acc[n][p];
    }
    __syncthreads();
    if (wave < 4) {   // waves 0-3 fold their partial into the stored one
        #pragma unroll
        for (int n = 0; n < 8; ++n)
            #pragma unroll
            for (int p = 0; p < 8; ++p) {
                const int e = wave * 2048 + (p + half * 8) * DIM + n * 16 + l15;
                buf[e] += acc[n][p];
            }
    }
    __syncthreads();

    // ---- final 4-way sum + normalization epilogue ----
    // H[i][c] = d_inv[i] * (sum + d_inv[i]*X[i][c])   (diag/self term of A+I)
    #pragma unroll
    for (int i = 0; i < 8; ++i) {
        const int e   = threadIdx.x * 8 + i;     // 0..2047
        const float s = buf[e] + buf[e + 2048] + buf[e + 4096] + buf[e + 6144];
        const int row = e >> 7, col = e & 127;
        const int gi  = r0 + row;
        const float di = d_inv[gi];
        H[(size_t)gi * DIM + col] = di * (s + di * X[(size_t)gi * DIM + col]);
    }
}

// ---------------------------------------------------------------------------
// Kernel 4: out = H @ W^T + b  via f32 WMMA 16x16x4 (full f32 precision).
// A frag (16x4 f32): lane l -> row l%16, K pair = (l/16)*2 + {0,1}.
// B frag (4x16 f32): lane l -> col l%16, same K pattern; W^T[k][c] = W[c][k].
// ---------------------------------------------------------------------------
__global__ void __launch_bounds__(256)
out_gemm(const float* __restrict__ H, const float* __restrict__ W,
         const float* __restrict__ bias, float* __restrict__ out) {
    const int lane = threadIdx.x & 31;
    const int wave = threadIdx.x >> 5;
    const int half = lane >> 4;
    const int l15  = lane & 15;
    const int wid  = blockIdx.x * 8 + wave;   // 0..4095
    const int r0   = (wid >> 3) * 16;
    const int c0   = (wid & 7) * 16;

    v8f acc = {};
#if __has_builtin(__builtin_amdgcn_wmma_f32_16x16x4_f32)
    const float* pH = H + (size_t)(r0 + l15) * DIM + half * 2;
    const float* pW = W + (size_t)(c0 + l15) * DIM + half * 2;
    #pragma unroll
    for (int k = 0; k < DIM; k += 4) {
        v2f a = *(const v2f*)(pH + k);
        v2f b = *(const v2f*)(pW + k);
        acc = __builtin_amdgcn_wmma_f32_16x16x4_f32(
            false, a, false, b, (short)0, acc, false, false);
    }
#else
    #pragma unroll
    for (int p = 0; p < 8; ++p) {
        const int i = r0 + p + half * 8;
        const int c = c0 + l15;
        float s = 0.f;
        for (int k = 0; k < DIM; ++k) s += H[(size_t)i * DIM + k] * W[(size_t)c * DIM + k];
        acc[p] = s;
    }
#endif
    #pragma unroll
    for (int p = 0; p < 8; ++p) {
        const int i = r0 + p + half * 8;
        const int c = c0 + l15;
        out[(size_t)i * DIM + c] = acc[p] + bias[c];
    }
}

// ---------------------------------------------------------------------------
extern "C" void kernel_launch(void* const* d_in, const int* in_sizes, int n_in,
                              void* d_out, int out_size, void* d_ws, size_t ws_size,
                              hipStream_t stream) {
    const float* X = (const float*)d_in[0];   // [8192,128]
    const float* A = (const float*)d_in[1];   // [8192,8192]
    const float* W = (const float*)d_in[2];   // [128,128]
    const float* b = (const float*)d_in[3];   // [128]
    float* out = (float*)d_out;               // [8192,128]

    char* ws = (char*)d_ws;
    float*  d_inv = (float*)ws;                                   // 32 KB
    __bf16* XsT   = (__bf16*)(ws + 32768);                        // 2 MB
    float*  H     = (float*)(ws + 32768 + (size_t)DIM * NROWS * sizeof(__bf16)); // 4 MB

    rowsum_rsqrt<<<NROWS, 256, 0, stream>>>(A, d_inv);
    build_xst<<<(NROWS * DIM) / 256, 256, 0, stream>>>(X, d_inv, XsT);
    spmm_wmma<<<NROWS / 16, 256, 0, stream>>>(A, X, XsT, d_inv, H);
    out_gemm<<<(NROWS / 16) * (DIM / 16) / 8, 256, 0, stream>>>(H, W, b, out);
}
